// Block_29832842838698
// MI455X (gfx1250) — matrix operand order repaired and checked by reading, not compiled, not run on hardware
//
#include <hip/hip_runtime.h>
#include <stdint.h>

// ---------------------------------------------------------------------------
// nGPT block for MI455X (gfx1250, wave32, WMMA).
// All GEMMs + attention matmuls run on v_wmma_f32_16x16x32_bf16.
// Weights are converted fp32->bf16 AND transposed once, so GEMM B-tiles are
// staged with contiguous b128 loads/stores (no scatter). GEMM main loop is
// software-pipelined with double-buffered LDS (1 barrier / K-step).
// ---------------------------------------------------------------------------

typedef __attribute__((ext_vector_type(16))) __bf16 v16bf;
typedef __attribute__((ext_vector_type(8)))  __bf16 v8bf;
typedef __attribute__((ext_vector_type(8)))  float  v8f;

#define WBATCH 2
#define WT 2048
#define WC 2048
#define WH 16
#define WD 128
#define BT (WBATCH * WT)   // 4096 token rows

__constant__ constexpr float BASE_SCALE    = 0.022097086912079608f;
__constant__ constexpr float SQK_MULT      = 45.254833995939045f;   // 1/BASE_SCALE
__constant__ constexpr float ALPHA_MULT    = 2.2627416997969522f;   // 0.05/BASE_SCALE
__constant__ constexpr float SUV_MULT      = 45.254833995939045f;   // sqrt(2048)
__constant__ constexpr float SOFTMAX_SCALE = 11.313708498984761f;   // sqrt(128)
__constant__ constexpr float INF_VAL       = 10000.0f;
__constant__ constexpr float EPSV          = 1.1920928955078125e-07f;

__device__ __forceinline__ __bf16 f2bf(float f) {
  uint32_t u = __builtin_bit_cast(uint32_t, f);
  u += 0x7FFFu + ((u >> 16) & 1u);             // round-to-nearest-even
  uint16_t s = (uint16_t)(u >> 16);
  return __builtin_bit_cast(__bf16, s);
}

__device__ __forceinline__ v8f wmma_bf16(const v16bf& a, const v16bf& b, const v8f& c) {
  return __builtin_amdgcn_wmma_f32_16x16x32_bf16(false, a, false, b, (short)0, c,
                                                 false, false);
}

// ---------------------------------------------------------------------------
// fp32 -> bf16 conversion (grid-stride)
// ---------------------------------------------------------------------------
__global__ void cvt_f32_bf16(const float* __restrict__ in, __bf16* __restrict__ out,
                             size_t n) {
  for (size_t i = (size_t)blockIdx.x * 256 + threadIdx.x; i < n;
       i += (size_t)gridDim.x * 256)
    out[i] = f2bf(in[i]);
}

// ---------------------------------------------------------------------------
// fp32 [R][Cc] -> bf16 transposed [Cc][R].  64x64 LDS-tiled, padded (+1) to
// stay bank-conflict-free; both global accesses fully coalesced.
// ---------------------------------------------------------------------------
__global__ __launch_bounds__(256)
void cvt_t_f32_bf16(const float* __restrict__ in, __bf16* __restrict__ out,
                    int R, int Cc) {
  __shared__ float tile[64][65];
  const int c0 = blockIdx.x * 64, r0 = blockIdx.y * 64;
  const int tx = threadIdx.x & 63, ty = threadIdx.x >> 6;   // 4 rows / pass
#pragma unroll
  for (int i = 0; i < 16; i++) {
    const int r = ty + i * 4;
    tile[r][tx] = in[(size_t)(r0 + r) * Cc + c0 + tx];
  }
  __syncthreads();
#pragma unroll
  for (int i = 0; i < 16; i++) {
    const int c = ty + i * 4;
    out[(size_t)(c0 + c) * R + r0 + tx] = f2bf(tile[tx][c]);
  }
}

// ---------------------------------------------------------------------------
// bf16 GEMM: C[M,N](f32) = A[M,K](bf16,row) x Bt[N,K](bf16, pre-transposed)
// 128x128 tile / workgroup, 8 waves (wave32), wave tile 32x64, K-step 32.
// Double-buffered LDS + register prefetch of the next K-tile; fragments are
// loaded before the 8 back-to-back WMMAs.
// ---------------------------------------------------------------------------
__global__ __launch_bounds__(256)
void gemm_bf16(const __bf16* __restrict__ A, const __bf16* __restrict__ Bt,
               float* __restrict__ Cm, int M, int N, int K) {
  __shared__ __bf16 As[2][128][40];   // [buf][m][k], padded
  __shared__ __bf16 Bs[2][128][40];   // [buf][n][k], padded

  const int tid  = threadIdx.x;
  const int wid  = tid >> 5, lane = tid & 31;
  const int hi   = lane >> 4, l16 = lane & 15;
  const int mbase = blockIdx.y * 128, nbase = blockIdx.x * 128;
  const int wm = (wid & 3) * 32, wn = (wid >> 2) * 64;

  const v8f z8 = {0.f, 0.f, 0.f, 0.f, 0.f, 0.f, 0.f, 0.f};
  v8f acc[2][4];
#pragma unroll
  for (int i = 0; i < 2; i++)
#pragma unroll
    for (int j = 0; j < 4; j++) acc[i][j] = z8;

  // staging: each thread stages half a row of A-tile and half a row of B-tile
  const int srow = tid >> 1, skoff = (tid & 1) * 16;
  const __bf16* aptr = A + (size_t)(mbase + srow) * K + skoff;
  const __bf16* bptr = Bt + (size_t)(nbase + srow) * K + skoff;

  // prologue: load K-step 0 into registers, store to buffer 0
  v8bf ra0 = *(const v8bf*)aptr;
  v8bf ra1 = *(const v8bf*)(aptr + 8);
  v8bf rb0 = *(const v8bf*)bptr;
  v8bf rb1 = *(const v8bf*)(bptr + 8);
  *(v8bf*)&As[0][srow][skoff]     = ra0;
  *(v8bf*)&As[0][srow][skoff + 8] = ra1;
  *(v8bf*)&Bs[0][srow][skoff]     = rb0;
  *(v8bf*)&Bs[0][srow][skoff + 8] = rb1;
  __syncthreads();

  const int nsteps = K >> 5;
  for (int s = 0; s < nsteps; ++s) {
    const int cur = s & 1;
    const int nxt = cur ^ 1;
    const bool has_next = (s + 1) < nsteps;

    // issue next K-tile global loads early (latency hidden behind WMMAs)
    if (has_next) {
      const __bf16* ap = aptr + (size_t)(s + 1) * 32;
      const __bf16* bp = bptr + (size_t)(s + 1) * 32;
      if (s + 2 < nsteps) {
        __builtin_prefetch(ap + 32, 0, 1);
        __builtin_prefetch(bp + 32, 0, 1);
      }
      ra0 = *(const v8bf*)ap;
      ra1 = *(const v8bf*)(ap + 8);
      rb0 = *(const v8bf*)bp;
      rb1 = *(const v8bf*)(bp + 8);
    }

    // load all fragments first, then issue 8 WMMAs back-to-back
    v16bf af[2], bfv[4];
#pragma unroll
    for (int s2 = 0; s2 < 2; s2++) {
      const int row = wm + s2 * 16 + l16;
      const v8bf lo = *(const v8bf*)&As[cur][row][hi * 16];
      const v8bf hb = *(const v8bf*)&As[cur][row][hi * 16 + 8];
#pragma unroll
      for (int e = 0; e < 8; e++) { af[s2][e] = lo[e]; af[s2][e + 8] = hb[e]; }
    }
#pragma unroll
    for (int n2 = 0; n2 < 4; n2++) {
      const int col = wn + n2 * 16 + l16;
      bfv[n2] = *(const v16bf*)&Bs[cur][col][hi * 16];
    }
#pragma unroll
    for (int n2 = 0; n2 < 4; n2++)
#pragma unroll
      for (int s2 = 0; s2 < 2; s2++)
        acc[s2][n2] = wmma_bf16(af[s2], bfv[n2], acc[s2][n2]);

    // stage next tile into the other buffer, single barrier per K-step
    if (has_next) {
      *(v8bf*)&As[nxt][srow][skoff]     = ra0;
      *(v8bf*)&As[nxt][srow][skoff + 8] = ra1;
      *(v8bf*)&Bs[nxt][srow][skoff]     = rb0;
      *(v8bf*)&Bs[nxt][srow][skoff + 8] = rb1;
    }
    __syncthreads();
  }

  // epilogue: C/D layout lane=N, vgpr v -> M = v + 8*hi
#pragma unroll
  for (int s2 = 0; s2 < 2; s2++)
#pragma unroll
    for (int n2 = 0; n2 < 4; n2++) {
      const int mg = mbase + wm + s2 * 16 + 8 * hi;
      const int ng = nbase + wn + n2 * 16 + l16;
      float* dst = Cm + (size_t)mg * N + ng;
#pragma unroll
      for (int v = 0; v < 8; v++) dst[(size_t)v * N] = acc[s2][n2][v];
    }
}

// ---------------------------------------------------------------------------
// RoPE + justnorm + sqk scaling; one wave per (b,t,h) row of D=128.
// Emits Q,K bf16 [B,H,T,D] (SOFTMAX_SCALE folded into Q) and V^T bf16 [B,H,D,T].
// ---------------------------------------------------------------------------
__global__ __launch_bounds__(256)
void rope_norm_qkv(const float* __restrict__ qf, const float* __restrict__ kf,
                   const float* __restrict__ vf, const float* __restrict__ sqk,
                   __bf16* __restrict__ Qb, __bf16* __restrict__ Kb,
                   __bf16* __restrict__ Vtb) {
  const int gw   = blockIdx.x * 8 + (threadIdx.x >> 5);
  const int lane = threadIdx.x & 31;
  const int h  = gw % WH;
  const int bt = gw / WH;
  const int t  = bt % WT;
  const int b  = bt / WT;

  const size_t src = (size_t)bt * WC + h * WD + lane * 4;
  float q0 = qf[src], q1 = qf[src + 1], q2 = qf[src + 2], q3 = qf[src + 3];
  float k0 = kf[src], k1 = kf[src + 1], k2 = kf[src + 2], k3 = kf[src + 3];
  float v0 = vf[src], v1 = vf[src + 1], v2 = vf[src + 2], v3 = vf[src + 3];

  // inv_freq[p] = 10000^(-(2p)/D), pairs (2p, 2p+1)
  const float LN10K = 9.210340371976184f;
  const int p0 = lane * 2, p1 = lane * 2 + 1;
  const float f0 = __expf(-LN10K * (float)(2 * p0) * (1.0f / WD));
  const float f1 = __expf(-LN10K * (float)(2 * p1) * (1.0f / WD));
  float c0, s0, c1, s1;
  __sincosf((float)t * f0, &s0, &c0);
  __sincosf((float)t * f1, &s1, &c1);

  const float qr0 = q0 * c0 - q1 * s0, qr1 = q1 * c0 + q0 * s0;
  const float qr2 = q2 * c1 - q3 * s1, qr3 = q3 * c1 + q2 * s1;
  const float kr0 = k0 * c0 - k1 * s0, kr1 = k1 * c0 + k0 * s0;
  const float kr2 = k2 * c1 - k3 * s1, kr3 = k3 * c1 + k2 * s1;

  float ssq = qr0 * qr0 + qr1 * qr1 + qr2 * qr2 + qr3 * qr3;
  float ssk = kr0 * kr0 + kr1 * kr1 + kr2 * kr2 + kr3 * kr3;
#pragma unroll
  for (int d = 1; d < 32; d <<= 1) {
    ssq += __shfl_xor(ssq, d, 32);
    ssk += __shfl_xor(ssk, d, 32);
  }
  const float iq = 1.0f / (sqrtf(ssq) + EPSV);
  const float ik = 1.0f / (sqrtf(ssk) + EPSV);

  const float* sc = sqk + h * WD + lane * 4;
  const float m0 = sc[0] * SQK_MULT, m1 = sc[1] * SQK_MULT;
  const float m2 = sc[2] * SQK_MULT, m3 = sc[3] * SQK_MULT;

  const size_t dq = ((size_t)(b * WH + h) * WT + t) * WD + lane * 4;
  Qb[dq]     = f2bf(m0 * qr0 * iq * SOFTMAX_SCALE);
  Qb[dq + 1] = f2bf(m1 * qr1 * iq * SOFTMAX_SCALE);
  Qb[dq + 2] = f2bf(m2 * qr2 * iq * SOFTMAX_SCALE);
  Qb[dq + 3] = f2bf(m3 * qr3 * iq * SOFTMAX_SCALE);
  Kb[dq]     = f2bf(m0 * kr0 * ik);
  Kb[dq + 1] = f2bf(m1 * kr1 * ik);
  Kb[dq + 2] = f2bf(m2 * kr2 * ik);
  Kb[dq + 3] = f2bf(m3 * kr3 * ik);

  const size_t dv = ((size_t)(b * WH + h) * WD + lane * 4) * WT + t;
  Vtb[dv]          = f2bf(v0);
  Vtb[dv + WT]     = f2bf(v1);
  Vtb[dv + 2 * WT] = f2bf(v2);
  Vtb[dv + 3 * WT] = f2bf(v3);
}

// ---------------------------------------------------------------------------
// Causal flash attention, one wave per 16-row q tile. Per 32-key block:
// 8 WMMAs for S=Q.K^T, online softmax on accumulator layout, P bounced
// through per-wave LDS to re-fragment, 8 WMMAs for O+=P.V.
// Y written as bf16 [B,T,C] (ready to be GEMM A-matrix for Wo).
// ---------------------------------------------------------------------------
__global__ __launch_bounds__(256)
void flash_attn(const __bf16* __restrict__ Qb, const __bf16* __restrict__ Kb,
                const __bf16* __restrict__ Vtb, __bf16* __restrict__ Yb) {
  __shared__ __bf16 Ps[8][16][40];   // per-wave 16x32 P tile (padded)

  const int bh  = blockIdx.x;
  const int b   = bh >> 4, h = bh & 15;
  const int wid = threadIdx.x >> 5, lane = threadIdx.x & 31;
  const int hi  = lane >> 4, l16 = lane & 15;
  const int qbase = blockIdx.y * 128 + wid * 16;

  const __bf16* Qp = Qb + (size_t)bh * WT * WD;
  const __bf16* Kp = Kb + (size_t)bh * WT * WD;
  const __bf16* Vp = Vtb + (size_t)bh * WD * WT;

  // Q fragments (resident for whole key loop)
  v16bf aq[4];
  {
    const __bf16* qr = Qp + (size_t)(qbase + l16) * WD;
#pragma unroll
    for (int kk = 0; kk < 4; kk++) {
      const v8bf lo = *(const v8bf*)(qr + kk * 32 + hi * 16);
      const v8bf hb = *(const v8bf*)(qr + kk * 32 + hi * 16 + 8);
#pragma unroll
      for (int e = 0; e < 8; e++) { aq[kk][e] = lo[e]; aq[kk][e + 8] = hb[e]; }
    }
  }

  const v8f z8 = {0.f, 0.f, 0.f, 0.f, 0.f, 0.f, 0.f, 0.f};
  v8f o[8];
#pragma unroll
  for (int nt = 0; nt < 8; nt++) o[nt] = z8;
  float rmax[8], rsum[8];
#pragma unroll
  for (int v = 0; v < 8; v++) { rmax[v] = -3.0e38f; rsum[v] = 0.f; }

  const int kblocks = (qbase + 15) / 32 + 1;   // causal: last key <= qbase+15
  for (int kb = 0; kb < kblocks; ++kb) {
    const int kbase = kb * 32;
    // ---- S = Q.K^T for 32 keys (two 16x16 tiles) ----
    v8f s0 = z8, s1 = z8;
    const __bf16* kr0 = Kp + (size_t)(kbase + l16) * WD + hi * 16;
    const __bf16* kr1 = kr0 + (size_t)16 * WD;
#pragma unroll
    for (int kk = 0; kk < 4; kk++) {
      const v16bf b0 = *(const v16bf*)(kr0 + kk * 32);
      const v16bf b1 = *(const v16bf*)(kr1 + kk * 32);
      s0 = wmma_bf16(aq[kk], b0, s0);
      s1 = wmma_bf16(aq[kk], b1, s1);
    }
    // ---- mask + online softmax (accumulator layout: lane=key, vgpr=m) ----
#pragma unroll
    for (int v = 0; v < 8; v++) {
      const int m    = v + 8 * hi;
      const int qi   = qbase + m;
      const int key0 = kbase + l16;
      const int key1 = kbase + 16 + l16;
      const float x0 = (key0 <= qi) ? s0[v] : -INF_VAL;
      const float x1 = (key1 <= qi) ? s1[v] : -INF_VAL;
      float bm = fmaxf(x0, x1);
#pragma unroll
      for (int d = 1; d < 16; d <<= 1) bm = fmaxf(bm, __shfl_xor(bm, d, 32));
      const float nm = fmaxf(rmax[v], bm);
      const float al = __expf(rmax[v] - nm);
      const float p0 = __expf(x0 - nm);
      const float p1 = __expf(x1 - nm);
      float ps = p0 + p1;
#pragma unroll
      for (int d = 1; d < 16; d <<= 1) ps += __shfl_xor(ps, d, 32);
      rsum[v] = rsum[v] * al + ps;
      rmax[v] = nm;
#pragma unroll
      for (int nt = 0; nt < 8; nt++) o[nt][v] *= al;
      Ps[wid][m][l16]      = f2bf((key0 <= qi) ? p0 : 0.f);   // where(mask,0)
      Ps[wid][m][16 + l16] = f2bf((key1 <= qi) ? p1 : 0.f);
    }
    // ---- P as A-fragment (same wave, LDS ops in-order) ----
    v16bf ap;
    {
      const v8bf lo = *(const v8bf*)&Ps[wid][l16][hi * 16];
      const v8bf hb = *(const v8bf*)&Ps[wid][l16][hi * 16 + 8];
#pragma unroll
      for (int e = 0; e < 8; e++) { ap[e] = lo[e]; ap[e + 8] = hb[e]; }
    }
    // ---- O += P.V (V^T rows contiguous along keys) ----
#pragma unroll
    for (int nt = 0; nt < 8; nt++) {
      const v16bf bv =
          *(const v16bf*)(Vp + (size_t)(nt * 16 + l16) * WT + kbase + hi * 16);
      o[nt] = wmma_bf16(ap, bv, o[nt]);
    }
  }

  // ---- finalize + store Y bf16 at [b, t, h*D + d] ----
  float inv[8];
#pragma unroll
  for (int v = 0; v < 8; v++) inv[v] = 1.0f / rsum[v];
#pragma unroll
  for (int nt = 0; nt < 8; nt++)
#pragma unroll
    for (int v = 0; v < 8; v++) {
      const int m = v + 8 * hi;
      Yb[(size_t)(b * WT + qbase + m) * WC + h * WD + nt * 16 + l16] =
          f2bf(o[nt][v] * inv[v]);
    }
}

// ---------------------------------------------------------------------------
// h = justnorm(justnorm(h) + |alpha*ALPHA_MULT| * (justnorm(d) - justnorm(h)))
// One 256-thread block per row of C=2048.
// ---------------------------------------------------------------------------
__global__ __launch_bounds__(256)
void residual_norm(const float* __restrict__ hin, const float* __restrict__ dres,
                   const float* __restrict__ alpha, float* __restrict__ outf,
                   __bf16* __restrict__ outb, int write_bf) {
  __shared__ float redA[8], redB[8];
  const int row = blockIdx.x;
  const int tid = threadIdx.x;
  const int wid = tid >> 5, lane = tid & 31;
  const float* hr = hin + (size_t)row * WC;
  const float* dr = dres + (size_t)row * WC;

  float hv[8], dv[8];
  float ssh = 0.f, ssd = 0.f;
#pragma unroll
  for (int i = 0; i < 8; i++) {
    const int j = tid + i * 256;
    hv[i] = hr[j]; dv[i] = dr[j];
    ssh += hv[i] * hv[i]; ssd += dv[i] * dv[i];
  }
#pragma unroll
  for (int d = 1; d < 32; d <<= 1) {
    ssh += __shfl_xor(ssh, d, 32);
    ssd += __shfl_xor(ssd, d, 32);
  }
  if (lane == 0) { redA[wid] = ssh; redB[wid] = ssd; }
  __syncthreads();
  ssh = 0.f; ssd = 0.f;
#pragma unroll
  for (int w = 0; w < 8; w++) { ssh += redA[w]; ssd += redB[w]; }
  const float ih = 1.f / (sqrtf(ssh) + EPSV);
  const float id = 1.f / (sqrtf(ssd) + EPSV);

  float tv[8], sst = 0.f;
#pragma unroll
  for (int i = 0; i < 8; i++) {
    const int j = tid + i * 256;
    const float nh = hv[i] * ih, nd = dv[i] * id;
    const float lr = fabsf(alpha[j] * ALPHA_MULT);
    const float tt = nh + lr * (nd - nh);
    tv[i] = tt; sst += tt * tt;
  }
#pragma unroll
  for (int d = 1; d < 32; d <<= 1) sst += __shfl_xor(sst, d, 32);
  __syncthreads();
  if (lane == 0) redA[wid] = sst;
  __syncthreads();
  sst = 0.f;
#pragma unroll
  for (int w = 0; w < 8; w++) sst += redA[w];
  const float it = 1.f / (sqrtf(sst) + EPSV);

#pragma unroll
  for (int i = 0; i < 8; i++) {
    const int j = tid + i * 256;
    const float ov = tv[i] * it;
    outf[(size_t)row * WC + j] = ov;
    if (write_bf) outb[(size_t)row * WC + j] = f2bf(ov);
  }
}

// ---------------------------------------------------------------------------
// x_mlp = (uv[:, :4C]*suv_u*SUV_MULT) * silu(uv[:, 4C:]*suv_g*SUV_MULT) -> bf16
// ---------------------------------------------------------------------------
__global__ void uv_silu(const float* __restrict__ uv, const float* __restrict__ suv,
                        __bf16* __restrict__ x) {
  const size_t n = (size_t)BT * 8192;
  for (size_t i = (size_t)blockIdx.x * 256 + threadIdx.x; i < n;
       i += (size_t)gridDim.x * 256) {
    const size_t row = i >> 13;
    const int j = (int)(i & 8191);
    const float u = uv[row * 16384 + j] * (suv[j] * SUV_MULT);
    const float g = uv[row * 16384 + 8192 + j] * (suv[8192 + j] * SUV_MULT);
    const float s = g / (1.f + __expf(-g));
    x[i] = f2bf(u * s);
  }
}

// ---------------------------------------------------------------------------
// Host-side orchestration
// ---------------------------------------------------------------------------
extern "C" void kernel_launch(void* const* d_in, const int* in_sizes, int n_in,
                              void* d_out, int out_size, void* d_ws, size_t ws_size,
                              hipStream_t stream) {
  (void)in_sizes; (void)n_in; (void)out_size; (void)ws_size;
  const float* h      = (const float*)d_in[0];
  /* d_in[1] = mask (causal, hardcoded) */
  const float* Wq     = (const float*)d_in[2];
  const float* Wk     = (const float*)d_in[3];
  const float* Wv     = (const float*)d_in[4];
  const float* Wo     = (const float*)d_in[5];
  const float* Wfc    = (const float*)d_in[6];
  const float* Wproj  = (const float*)d_in[7];
  const float* sqk    = (const float*)d_in[8];
  const float* suv    = (const float*)d_in[9];
  const float* aalpha = (const float*)d_in[10];
  const float* malpha = (const float*)d_in[11];
  float* out = (float*)d_out;

  char* ws = (char*)d_ws;
  size_t off = 0;
  auto alloc = [&](size_t bytes) -> char* {
    char* p = ws + off;
    off += (bytes + 255) & ~(size_t)255;
    return p;
  };

  __bf16* hbf     = (__bf16*)alloc((size_t)BT * WC * 2);
  __bf16* wqbf    = (__bf16*)alloc((size_t)WC * WC * 2);        // transposed [N][K]
  __bf16* wkbf    = (__bf16*)alloc((size_t)WC * WC * 2);
  __bf16* wvbf    = (__bf16*)alloc((size_t)WC * WC * 2);
  __bf16* wobf    = (__bf16*)alloc((size_t)WC * WC * 2);
  __bf16* wfcbf   = (__bf16*)alloc((size_t)WC * 8 * WC * 2);
  __bf16* wprojbf = (__bf16*)alloc((size_t)4 * WC * WC * 2);
  float*  qf      = (float*)alloc((size_t)BT * WC * 4);   // reused as h_att
  float*  kf      = (float*)alloc((size_t)BT * WC * 4);   // reused as h2 (f32)
  float*  vf      = (float*)alloc((size_t)BT * WC * 4);   // reused as h_mlp
  __bf16* qbf2    = (__bf16*)alloc((size_t)BT * WC * 2);
  __bf16* kbf2    = (__bf16*)alloc((size_t)BT * WC * 2);
  __bf16* vtbf    = (__bf16*)alloc((size_t)BT * WC * 2);
  __bf16* ybf     = (__bf16*)alloc((size_t)BT * WC * 2);
  __bf16* h2bf    = (__bf16*)alloc((size_t)BT * WC * 2);
  float*  uvf     = (float*)alloc((size_t)BT * 8 * WC * 4);
  __bf16* xmlpbf  = (__bf16*)alloc((size_t)BT * 4 * WC * 2);
  float*  hatt = qf;
  float*  h2f  = kf;
  float*  hmlp = vf;

  const dim3 blk(256);

  // activations: plain convert; weights: convert + transpose to [N][K]
  cvt_f32_bf16<<<4096, blk, 0, stream>>>(h, hbf, (size_t)BT * WC);
  cvt_t_f32_bf16<<<dim3(WC / 64, WC / 64), blk, 0, stream>>>(Wq, wqbf, WC, WC);
  cvt_t_f32_bf16<<<dim3(WC / 64, WC / 64), blk, 0, stream>>>(Wk, wkbf, WC, WC);
  cvt_t_f32_bf16<<<dim3(WC / 64, WC / 64), blk, 0, stream>>>(Wv, wvbf, WC, WC);
  cvt_t_f32_bf16<<<dim3(WC / 64, WC / 64), blk, 0, stream>>>(Wo, wobf, WC, WC);
  cvt_t_f32_bf16<<<dim3(8 * WC / 64, WC / 64), blk, 0, stream>>>(Wfc, wfcbf,
                                                                 WC, 8 * WC);
  cvt_t_f32_bf16<<<dim3(WC / 64, 4 * WC / 64), blk, 0, stream>>>(Wproj, wprojbf,
                                                                 4 * WC, WC);

  // QKV projections
  gemm_bf16<<<dim3(WC / 128, BT / 128), blk, 0, stream>>>(hbf, wqbf, qf, BT, WC, WC);
  gemm_bf16<<<dim3(WC / 128, BT / 128), blk, 0, stream>>>(hbf, wkbf, kf, BT, WC, WC);
  gemm_bf16<<<dim3(WC / 128, BT / 128), blk, 0, stream>>>(hbf, wvbf, vf, BT, WC, WC);

  // RoPE + norm + layouts for attention
  rope_norm_qkv<<<(BT * WH) / 8, blk, 0, stream>>>(qf, kf, vf, sqk, qbf2, kbf2, vtbf);

  // causal flash attention
  flash_attn<<<dim3(WBATCH * WH, WT / 128), blk, 0, stream>>>(qbf2, kbf2, vtbf, ybf);

  // output projection + attn residual
  gemm_bf16<<<dim3(WC / 128, BT / 128), blk, 0, stream>>>(ybf, wobf, hatt, BT, WC, WC);
  residual_norm<<<BT, blk, 0, stream>>>(h, hatt, aalpha, h2f, h2bf, 1);

  // MLP
  gemm_bf16<<<dim3(8 * WC / 128, BT / 128), blk, 0, stream>>>(h2bf, wfcbf, uvf,
                                                              BT, 8 * WC, WC);
  uv_silu<<<8192, blk, 0, stream>>>(uvf, suv, xmlpbf);
  gemm_bf16<<<dim3(WC / 128, BT / 128), blk, 0, stream>>>(xmlpbf, wprojbf, hmlp,
                                                          BT, WC, 4 * WC);
  residual_norm<<<BT, blk, 0, stream>>>(h2f, hmlp, malpha, out, nullptr, 0);
}
